// LinearAttention_30940944400685
// MI455X (gfx1250) — matrix operand order repaired and checked by reading, not compiled, not run on hardware
//
#include <hip/hip_runtime.h>
#include <hip/hip_bf16.h>
#include <stdint.h>

// ---------------------------------------------------------------------------
// LinearAttention on gfx1250 (MI455X): bf16 WMMA pipeline, f32 accumulate.
// Big GEMMs stage the shared weight tile in LDS via async-to-LDS copies
// (double buffered, ASYNCcnt-tracked) when the toolchain exposes the builtin.
// ---------------------------------------------------------------------------

typedef __attribute__((ext_vector_type(16))) __bf16 v16bf;
typedef __attribute__((ext_vector_type(8)))  float v8f;
typedef __attribute__((ext_vector_type(4)))  int   v4i;

#define TOK   32768      // total tokens (B*H*W)
#define NTOK  4096       // tokens per batch
#define NBAT  8
#define CCH   512
#define F3    1536
#define NHEAD 8
#define DH    64

#if defined(__has_builtin)
#if __has_builtin(__builtin_amdgcn_global_load_async_to_lds_b128)
#define ASYNC_LDS 1
#else
#define ASYNC_LDS 0
#endif
#else
#define ASYNC_LDS 0
#endif

static __device__ __forceinline__ unsigned short f2bf(float x) {
    union { float f; unsigned u; } v; v.f = x;
    unsigned r = v.u + 0x7FFFu + ((v.u >> 16) & 1u);   // round-to-nearest-even
    return (unsigned short)(r >> 16);
}

static __device__ __forceinline__ v8f vzero8() {
    v8f z = {0.f,0.f,0.f,0.f,0.f,0.f,0.f,0.f};
    return z;
}

union FragU { v16bf bf; unsigned u[8]; };

// A-matrix 16x32 bf16 fragment (row-major source, K contiguous).
// lanes 0-15 hold M=lane, K groups {0..7,16..23}; lanes 16-31 K {8..15,24..31};
// each VGPR = 2 packed K values.
static __device__ __forceinline__ v16bf load_a(const unsigned short* rowbase,
                                               int lane, int ks) {
    const unsigned* p = (const unsigned*)rowbase + ks * 16;
    int kb2 = (lane >> 4) * 4;
    FragU f;
    f.u[0] = p[kb2 + 0]; f.u[1] = p[kb2 + 1];
    f.u[2] = p[kb2 + 2]; f.u[3] = p[kb2 + 3];
    f.u[4] = p[kb2 + 8]; f.u[5] = p[kb2 + 9];
    f.u[6] = p[kb2 + 10]; f.u[7] = p[kb2 + 11];
    return f.bf;
}

// B-matrix 32x16 bf16 fragment (column-major source: column n stored with
// K contiguous). lanes 0-15: K {0..7,16..23}, lanes 16-31: K {8..15,24..31}, col = lane%16.
static __device__ __forceinline__ v16bf load_b(const unsigned short* colbase,
                                               int lane, int ks) {
    const unsigned* p = (const unsigned*)colbase + ks * 16;
    int kb2 = (lane >> 4) * 4;
    FragU f;
#pragma unroll
    for (int r = 0; r < 8; ++r) f.u[r] = p[kb2 + r + (r >> 2) * 4];
    return f.bf;
}

static __device__ __forceinline__ v8f wmma_bf16(v16bf a, v16bf b, v8f c) {
    return __builtin_amdgcn_wmma_f32_16x16x32_bf16(false, a, false, b,
                                                   (short)0, c, false, false);
}

static __device__ __forceinline__ void wait_async_copies() {
#if ASYNC_LDS
#if __has_builtin(__builtin_amdgcn_s_wait_asynccnt)
    __builtin_amdgcn_s_wait_asynccnt(0);
#else
    asm volatile("s_wait_asynccnt 0x0" ::: "memory");
#endif
#endif
}

// Copy one 64(N) x 32(K) bf16 weight tile (4KB) into LDS; one b128 per thread.
static __device__ __forceinline__ void copy_b_tile(const unsigned short* __restrict__ Bt,
                                                   unsigned short* dst,
                                                   int n0, int ks, int tid) {
    int c  = tid >> 2;          // column 0..63
    int kq = (tid & 3) << 3;    // k offset 0,8,16,24
    const unsigned short* src = Bt + (size_t)(n0 + c) * CCH + ks * 32 + kq;
#if ASYNC_LDS
    __builtin_amdgcn_global_load_async_to_lds_b128(
        (__attribute__((address_space(1))) v4i*)src,
        (__attribute__((address_space(3))) v4i*)(dst + c * 32 + kq),
        0, 0);
#else
    *(uint4*)(dst + c * 32 + kq) = *(const uint4*)src;
#endif
}

// ---------------------------------------------------------------------------
// conversion kernels
// ---------------------------------------------------------------------------
__global__ void k_cvt(const float* __restrict__ src, unsigned short* __restrict__ dst, int n) {
    int i = blockIdx.x * blockDim.x + threadIdx.x;
    if (i < n) dst[i] = f2bf(src[i]);
}

// src [K][N] f32 -> dst [N][K] bf16 (so GEMM B columns have contiguous K)
__global__ void k_cvt_t(const float* __restrict__ src, unsigned short* __restrict__ dst,
                        int K, int N) {
    int i = blockIdx.x * blockDim.x + threadIdx.x;
    if (i < K * N) {
        int k = i / N, n = i - k * N;
        dst[(size_t)n * K + k] = f2bf(src[i]);
    }
}

// ---------------------------------------------------------------------------
// Weight GEMM: C[M,ldC] (+bias) = A[M,512](bf16) @ Bt'[N,512](bf16)
// grid (M/128, N/64), block 256 (8 waves); wave => 16x64 strip.
// B tile double-buffered in LDS via async-to-LDS; A streamed from global.
// ---------------------------------------------------------------------------
__global__ void __launch_bounds__(256) k_gemm_nt(const unsigned short* __restrict__ A,
                                                 const unsigned short* __restrict__ Bt,
                                                 float* __restrict__ C, int ldC,
                                                 const float* __restrict__ bias) {
    __shared__ unsigned short ldsB[2][64 * 32];
    int tid = threadIdx.x;
    int lane = tid & 31, w = tid >> 5;
    int m0 = blockIdx.x * 128 + w * 16;
    int n0 = blockIdx.y * 64;
    const unsigned short* arow = A + (size_t)(m0 + (lane & 15)) * CCH;

    copy_b_tile(Bt, ldsB[0], n0, 0, tid);
    wait_async_copies();
    __syncthreads();

    v8f acc[4] = {vzero8(), vzero8(), vzero8(), vzero8()};
    for (int ks = 0; ks < CCH / 32; ++ks) {
        int cur = ks & 1;
        if (ks + 1 < CCH / 32)
            copy_b_tile(Bt, ldsB[cur ^ 1], n0, ks + 1, tid);
        v16bf a = load_a(arow, lane, ks);
#pragma unroll
        for (int t = 0; t < 4; ++t) {
            const unsigned short* bcol = &ldsB[cur][(t * 16 + (lane & 15)) * 32];
            acc[t] = wmma_bf16(a, load_b(bcol, lane, 0), acc[t]);
        }
        if (ks + 1 < CCH / 32) {
            wait_async_copies();
            __syncthreads();
        }
    }
    int mr = m0 + ((lane >> 4) << 3);
    int nc = lane & 15;
#pragma unroll
    for (int t = 0; t < 4; ++t)
#pragma unroll
        for (int r = 0; r < 8; ++r) {
            int cc = n0 + t * 16 + nc;
            float v = acc[t][r];
            if (bias) v += bias[cc];
            C[(size_t)(mr + r) * ldC + cc] = v;
        }
}

// ---------------------------------------------------------------------------
// q softmax over d (64) per (token, head); writes bf16 token-major, * 1/8 scale
// grid TOK, block 256: wave = head
// ---------------------------------------------------------------------------
__global__ void __launch_bounds__(256) k_softmax_q(const float* __restrict__ qkv,
                                                   unsigned short* __restrict__ qb) {
    int token = blockIdx.x;
    int head = threadIdx.x >> 5;
    int lane = threadIdx.x & 31;
    const float* base = qkv + (size_t)token * F3 + head * DH;
    float2 v = ((const float2*)base)[lane];               // d = 2*lane, 2*lane+1
    float m = fmaxf(v.x, v.y);
#pragma unroll
    for (int o = 16; o > 0; o >>= 1) m = fmaxf(m, __shfl_xor(m, o, 32));
    float e0 = __expf(v.x - m), e1 = __expf(v.y - m);
    float s = e0 + e1;
#pragma unroll
    for (int o = 16; o > 0; o >>= 1) s += __shfl_xor(s, o, 32);
    float inv = 0.125f / s;                               // fold scale = DH^-0.5
    unsigned pack = (unsigned)f2bf(e0 * inv) | ((unsigned)f2bf(e1 * inv) << 16);
    unsigned short* ob = qb + (size_t)token * CCH + head * DH;
    ((unsigned*)ob)[lane] = pack;
}

// ---------------------------------------------------------------------------
// k softmax over n (4096) + transpose -> kT bf16 [bh][d][n]
// v transpose                        -> vT bf16 [bh][e][n]
// grid 8192: first 4096 = k rows, rest = v rows.  block 256, 16 tokens/thread.
// ---------------------------------------------------------------------------
__global__ void __launch_bounds__(256) k_prep_kv(const float* __restrict__ qkv,
                                                 unsigned short* __restrict__ kT,
                                                 unsigned short* __restrict__ vT) {
    int bid = blockIdx.x;
    int isV = bid >> 12;
    int id = bid & 4095;
    int b = id >> 9, rem = id & 511, h = rem >> 6, d = rem & 63;
    int tid = threadIdx.x;
    int f = (isV ? 2 * CCH : CCH) + h * DH + d;
    const float* src = qkv + (size_t)b * NTOK * F3 + f;
    float val[16];
#pragma unroll
    for (int j = 0; j < 16; ++j) val[j] = src[(size_t)(tid + j * 256) * F3];
    size_t obase = (size_t)((b * NHEAD + h) * DH + d) * NTOK;
    if (isV) {
#pragma unroll
        for (int j = 0; j < 16; ++j) vT[obase + tid + j * 256] = f2bf(val[j]);
        return;
    }
    __shared__ float shm[8], shs[8];
    int wv = tid >> 5, ln = tid & 31;
    float m = val[0];
#pragma unroll
    for (int j = 1; j < 16; ++j) m = fmaxf(m, val[j]);
#pragma unroll
    for (int o = 16; o > 0; o >>= 1) m = fmaxf(m, __shfl_xor(m, o, 32));
    if (ln == 0) shm[wv] = m;
    __syncthreads();
    m = shm[0];
#pragma unroll
    for (int i = 1; i < 8; ++i) m = fmaxf(m, shm[i]);
    float s = 0.f;
#pragma unroll
    for (int j = 0; j < 16; ++j) { val[j] = __expf(val[j] - m); s += val[j]; }
#pragma unroll
    for (int o = 16; o > 0; o >>= 1) s += __shfl_xor(s, o, 32);
    if (ln == 0) shs[wv] = s;
    __syncthreads();
    s = 0.f;
#pragma unroll
    for (int i = 0; i < 8; ++i) s += shs[i];
    float inv = 1.f / s;
#pragma unroll
    for (int j = 0; j < 16; ++j) kT[obase + tid + j * 256] = f2bf(val[j] * inv);
}

// ---------------------------------------------------------------------------
// ctx GEMM: ctx[e][d] = sum_n kT[d][n] * vT[e][n]   per (b,h)
// grid 64 (=bh), block 256: waves 0-3 / 4-7 split K=4096 in halves,
// LDS-combine, waves 0-3 store transposed bf16 ctx [bh][e*64+d].
// ---------------------------------------------------------------------------
__global__ void __launch_bounds__(256) k_gemm_ctx(const unsigned short* __restrict__ kT,
                                                  const unsigned short* __restrict__ vT,
                                                  unsigned short* __restrict__ ctx) {
    int bh = blockIdx.x;
    int lane = threadIdx.x & 31, w = threadIdx.x >> 5;
    int m0 = (w & 3) * 16;          // d-tile
    int kh = w >> 2;                // K half
    const unsigned short* Ab = kT + (size_t)bh * DH * NTOK;
    const unsigned short* Bb = vT + (size_t)bh * DH * NTOK;
    const unsigned short* arow = Ab + (size_t)(m0 + (lane & 15)) * NTOK;
    const unsigned short* bcol[4];
#pragma unroll
    for (int t = 0; t < 4; ++t)
        bcol[t] = Bb + (size_t)(t * 16 + (lane & 15)) * NTOK;
    v8f acc[4] = {vzero8(), vzero8(), vzero8(), vzero8()};
    for (int ks = kh * 64; ks < kh * 64 + 64; ++ks) {
        v16bf a = load_a(arow, lane, ks);
#pragma unroll
        for (int t = 0; t < 4; ++t)
            acc[t] = wmma_bf16(a, load_b(bcol[t], lane, ks), acc[t]);
    }
    __shared__ float lds[4096];
    if (kh == 1) {
#pragma unroll
        for (int t = 0; t < 4; ++t)
#pragma unroll
            for (int r = 0; r < 8; ++r)
                lds[((w - 4) * 4 + t) * 256 + lane * 8 + r] = acc[t][r];
    }
    __syncthreads();
    if (kh == 0) {
        unsigned short* cb = ctx + (size_t)bh * (DH * DH);
#pragma unroll
        for (int t = 0; t < 4; ++t) {
            int e = t * 16 + (lane & 15);
#pragma unroll
            for (int r = 0; r < 8; ++r) {
                int d = m0 + r + ((lane >> 4) << 3);
                float vsum = acc[t][r] + lds[(w * 4 + t) * 256 + lane * 8 + r];
                cb[e * DH + d] = f2bf(vsum);
            }
        }
    }
}

// ---------------------------------------------------------------------------
// attn-out GEMM: out[n][e] = sum_d qb[n][d] * ctx[e][d] per (b,h) -> bf16
// grid (32, 64): blockIdx.x tiles 128 tokens, blockIdx.y = bh
// ---------------------------------------------------------------------------
__global__ void __launch_bounds__(256) k_gemm_attnout(const unsigned short* __restrict__ qb,
                                                      const unsigned short* __restrict__ ctx,
                                                      unsigned short* __restrict__ oat) {
    int lane = threadIdx.x & 31, w = threadIdx.x >> 5;
    int bh = blockIdx.y, b = bh >> 3, h = bh & 7;
    int m0 = blockIdx.x * 128 + w * 16;           // token within batch
    size_t tokbase = (size_t)b * NTOK;
    const unsigned short* arow = qb + (tokbase + m0 + (lane & 15)) * CCH + h * DH;
    const unsigned short* bcol[4];
#pragma unroll
    for (int t = 0; t < 4; ++t)
        bcol[t] = ctx + (size_t)bh * (DH * DH) + (size_t)(t * 16 + (lane & 15)) * DH;
    v8f acc[4] = {vzero8(), vzero8(), vzero8(), vzero8()};
    for (int ks = 0; ks < DH / 32; ++ks) {
        v16bf a = load_a(arow, lane, ks);
#pragma unroll
        for (int t = 0; t < 4; ++t)
            acc[t] = wmma_bf16(a, load_b(bcol[t], lane, ks), acc[t]);
    }
    int mr = m0 + ((lane >> 4) << 3);
    int nc = lane & 15;
#pragma unroll
    for (int t = 0; t < 4; ++t)
#pragma unroll
        for (int r = 0; r < 8; ++r)
            oat[(tokbase + mr + r) * CCH + h * DH + t * 16 + nc] = f2bf(acc[t][r]);
}

// ---------------------------------------------------------------------------
// LayerNorm over 512 channels per token (biased variance), f32 out
// grid TOK, block 256 (2 channels/thread)
// ---------------------------------------------------------------------------
__global__ void __launch_bounds__(256) k_ln(const float* __restrict__ y,
                                            const float* __restrict__ g,
                                            const float* __restrict__ bta,
                                            float* __restrict__ out) {
    int token = blockIdx.x, tid = threadIdx.x;
    const float* row = y + (size_t)token * CCH;
    float a = row[tid], c = row[tid + 256];
    float s = a + c, q = a * a + c * c;
#pragma unroll
    for (int o = 16; o > 0; o >>= 1) {
        s += __shfl_xor(s, o, 32);
        q += __shfl_xor(q, o, 32);
    }
    __shared__ float ss[8], sq[8];
    int wv = tid >> 5, ln = tid & 31;
    if (ln == 0) { ss[wv] = s; sq[wv] = q; }
    __syncthreads();
    s = 0.f; q = 0.f;
#pragma unroll
    for (int i = 0; i < 8; ++i) { s += ss[i]; q += sq[i]; }
    float mean = s * (1.f / 512.f);
    float var = q * (1.f / 512.f) - mean * mean;
    float inv = rsqrtf(var + 1e-5f);
    float* orow = out + (size_t)token * CCH;
    orow[tid]       = (a - mean) * inv * g[tid] + bta[tid];
    orow[tid + 256] = (c - mean) * inv * g[tid + 256] + bta[tid + 256];
}

// ---------------------------------------------------------------------------
// host launcher
// ---------------------------------------------------------------------------
extern "C" void kernel_launch(void* const* d_in, const int* in_sizes, int n_in,
                              void* d_out, int out_size, void* d_ws, size_t ws_size,
                              hipStream_t stream) {
    const float* x     = (const float*)d_in[0];
    const float* w_qkv = (const float*)d_in[1];
    const float* w_out = (const float*)d_in[2];
    const float* b_out = (const float*)d_in[3];
    const float* ln_g  = (const float*)d_in[4];
    const float* ln_b  = (const float*)d_in[5];

    char* ws = (char*)d_ws;
    // workspace layout (bytes); qkv region is reused for oat + y after softmax.
    float*          qkv = (float*)ws;                              // 192 MB
    unsigned short* oat = (unsigned short*)ws;                     // 32 MB (alias, later)
    float*          y   = (float*)(ws + 33554432);                 // 64 MB (alias, later)
    unsigned short* xb  = (unsigned short*)(ws + 201326592);       // 32 MB
    unsigned short* wqt = (unsigned short*)(ws + 234881024);       // 1.5 MB
    unsigned short* wot = (unsigned short*)(ws + 236453888);       // 0.5 MB
    unsigned short* qb  = (unsigned short*)(ws + 236978176);       // 32 MB
    unsigned short* kT  = (unsigned short*)(ws + 270532608);       // 32 MB
    unsigned short* vT  = (unsigned short*)(ws + 304087040);       // 32 MB
    unsigned short* ctx = (unsigned short*)(ws + 337641472);       // 0.5 MB
    // total ~ 338.2 MB

    // 1) bf16 conversions (weights transposed so GEMM-B columns are K-contig)
    k_cvt  <<<(TOK * CCH + 255) / 256, 256, 0, stream>>>(x, xb, TOK * CCH);
    k_cvt_t<<<(CCH * F3 + 255) / 256, 256, 0, stream>>>(w_qkv, wqt, CCH, F3);
    k_cvt_t<<<(CCH * CCH + 255) / 256, 256, 0, stream>>>(w_out, wot, CCH, CCH);

    // 2) qkv projection (LDS-staged async B tiles)
    k_gemm_nt<<<dim3(TOK / 128, F3 / 64), 256, 0, stream>>>(xb, wqt, qkv, F3, nullptr);

    // 3) softmaxes + transposes (consume qkv)
    k_softmax_q<<<TOK, 256, 0, stream>>>(qkv, qb);
    k_prep_kv  <<<8192, 256, 0, stream>>>(qkv, kT, vT);

    // 4) context = k @ v^T per head
    k_gemm_ctx<<<NBAT * NHEAD, 256, 0, stream>>>(kT, vT, ctx);

    // 5) attention output = q @ ctx (oat aliases dead qkv region)
    k_gemm_attnout<<<dim3(NTOK / 128, NBAT * NHEAD), 256, 0, stream>>>(qb, ctx, oat);

    // 6) output projection + bias (LDS-staged async B tiles)
    k_gemm_nt<<<dim3(TOK / 128, CCH / 64), 256, 0, stream>>>(oat, wot, y, CCH, b_out);

    // 7) channel LayerNorm -> final f32 output
    k_ln<<<TOK, 256, 0, stream>>>(y, ln_g, ln_b, (float*)d_out);
}
